// Net_39659728011900
// MI455X (gfx1250) — matrix operand order repaired and checked by reading, not compiled, not run on hardware
//
#include <hip/hip_runtime.h>
#include <hip/hip_bf16.h>
#include <math.h>

// ---------------------------------------------------------------------------
// MI455X / gfx1250, wave32. All matrix work via v_wmma_f32_16x16x32_f16.
// ---------------------------------------------------------------------------

typedef __attribute__((ext_vector_type(16))) _Float16 v16h;
typedef __attribute__((ext_vector_type(8)))  _Float16 v8h;
typedef __attribute__((ext_vector_type(8)))  float    v8f;

#define BB    64
#define NN    1600
#define SS    512
#define C1    20
#define WIDTH 128
#define KTOT  (SS * C1)   // 10240

// workspace byte offsets (all 256-aligned)
#define OFF_ZDATA 0u        // f16 [64][1600]          204800 B
#define OFF_SW1T  204800u   // f16 [128][96]            24576 B
#define OFF_SW2T  229376u   // f16 [2560][128]         655360 B
#define OFF_ZBUF  884736u   // f16 [64][10240]        1310720 B
#define OFF_FBUFT 2195456u  // f16 [128][10240]       2621440 B
#define OFF_Z2    4816896u  // f32 [64][128]            32768 B

// fast reciprocal (single v_rcp_f32, no IEEE div expansion)
__device__ __forceinline__ float frcp(float x){ return __builtin_amdgcn_rcpf(x); }
__device__ __forceinline__ float silu_f(float x){
  return x * frcp(1.f + __expf(-x));
}

__device__ __forceinline__ v16h mk16(v8h lo, v8h hi){
  v16h r;
#pragma unroll
  for (int i = 0; i < 8; ++i){ r[i] = lo[i]; r[8 + i] = hi[i]; }
  return r;
}

__device__ __forceinline__ v8f wmma_f16(v16h a, v16h b, v8f c){
  return __builtin_amdgcn_wmma_f32_16x16x32_f16(false, a, false, b, (short)0, c, false, false);
}

// ---------------------------------------------------------------------------
// k0: dtype/layout prep.  zdata = f16(data);  sw1T[n][k] = s_w1[k][n] (k<80,
// zero-padded to 96);  sw2T[j][k] = s_w2[k][j].
// ---------------------------------------------------------------------------
__global__ void k0_prep(const float* __restrict__ data,
                        const float* __restrict__ s_w1,
                        const float* __restrict__ s_w2,
                        _Float16* __restrict__ zdata,
                        _Float16* __restrict__ sw1T,
                        _Float16* __restrict__ sw2T){
  const int i = blockIdx.x * blockDim.x + threadIdx.x;
  if (i < BB * NN) zdata[i] = (_Float16)data[i];
  if (i < 128 * 96){
    const int n = i / 96, k = i % 96;
    sw1T[i] = (k < 80) ? (_Float16)s_w1[k * 128 + n] : (_Float16)0.f;
  }
  if (i < 2560 * 128){
    const int j = i / 128, k = i % 128;
    sw2T[i] = (_Float16)s_w2[k * 2560 + j];
  }
}

// ---------------------------------------------------------------------------
// k1: per group-sample s: generate filt1 tile-by-tile in LDS (transposed,
// WMMA-B friendly), then z1 = silu( data @ filt1 / N ) via WMMA.
// Block = 256 threads = 8 waves -> 4 M-tiles x 2 N-tiles of the 64x32 output.
// ---------------------------------------------------------------------------
#define KC  320
#define KCP 328   // padded LDS row stride (halves); 656 B = 41*16, keeps b128 alignment

__global__ void __launch_bounds__(256)
k1_layer1(const _Float16* __restrict__ zdata,
          const float* __restrict__ v_last,
          const float* __restrict__ fw1, const float* __restrict__ fb1,
          const float* __restrict__ fw2, const float* __restrict__ fb2,
          const float* __restrict__ fw3, const float* __restrict__ fb3,
          _Float16* __restrict__ zbuf){
  __shared__ _Float16 filtBT[32][KCP];              // filt1^T tile: [out_chan][point]
  __shared__ float W1[400], W2[400], W3[400];
  __shared__ float B1[20], B2[20], B3[20];
  __shared__ float Hs[9];

  const int s   = blockIdx.x;
  const int tid = threadIdx.x;

  for (int i = tid; i < 400; i += 256){ W1[i] = fw1[i]; W2[i] = fw2[i]; W3[i] = fw3[i]; }
  if (tid < 20){ B1[tid] = fb1[tid]; B2[tid] = fb2[tid]; B3[tid] = fb3[tid]; }
  if (tid < 9){
    const float p = (tid < 8) ? 0.1f * v_last[s * 8 + tid] : 0.f;
    Hs[tid] = p + ((tid % 4 == 0) ? 1.f : 0.f);     // H = I + pert
  }
  // zero channels 20..31 once (never overwritten)
  for (int i = tid; i < 12 * KCP; i += 256) filtBT[20 + i / KCP][i % KCP] = (_Float16)0.f;
  __syncthreads();

  const int w = tid >> 5, lane = tid & 31;
  const int hh = lane >> 4, ln = lane & 15;
  const int mt = w & 3, nt = w >> 2;
  const _Float16* Arow     = zdata + (mt * 16 + ln) * NN;   // data row (batch m)
  const _Float16* BrowBase = &filtBT[nt * 16 + ln][0];      // filt1^T row (chan n)

  v8f acc = {};
  for (int chunk = 0; chunk < NN / KC; ++chunk){
    const int k0g = chunk * KC;
    // --- generate filt1 for KC grid points (scalar MLP, all 256 threads) ---
    for (int p = tid; p < KC; p += 256){
      const int n  = k0g + p;
      const int ii = n / 40, jj = n % 40;
      const float x0 = -1.f + (2.f / 39.f) * (float)ii;
      const float x1 = -1.f + (2.f / 39.f) * (float)jj;
      const float y0 = Hs[0] * x0 + Hs[1] * x1 + Hs[2];
      const float y1 = Hs[3] * x0 + Hs[4] * x1 + Hs[5];
      const float y2 = Hs[6] * x0 + Hs[7] * x1 + Hs[8];
      const float inv = frcp(y2);
      const float t0 = y0 * inv, t1 = y1 * inv;
      float e[20];
      float f = 1.f;
#pragma unroll
      for (int k = 0; k < 5; ++k){
        e[k]      = __sinf(t0 * f);  e[5 + k]  = __cosf(t0 * f);
        e[10 + k] = __sinf(t1 * f);  e[15 + k] = __cosf(t1 * f);
        f *= 2.f;
      }
      float h[20], g[20];
      for (int o = 0; o < 20; ++o){
        float a = B1[o];
#pragma unroll 4
        for (int k = 0; k < 20; ++k) a += e[k] * W1[k * 20 + o];
        h[o] = silu_f(a);
      }
      for (int o = 0; o < 20; ++o){
        float a = B2[o];
#pragma unroll 4
        for (int k = 0; k < 20; ++k) a += h[k] * W2[k * 20 + o];
        g[o] = silu_f(a);
      }
      for (int o = 0; o < 20; ++o){
        float a = B3[o];
#pragma unroll 4
        for (int k = 0; k < 20; ++k) a += g[k] * W3[k * 20 + o];
        filtBT[o][p] = (_Float16)a;
      }
    }
    __syncthreads();
    // --- WMMA accumulate over this K chunk (A: global b128, B: LDS b128) ---
#pragma unroll 5
    for (int ks = 0; ks < KC / 32; ++ks){
      const int kk = ks * 32;
      v8h alo = *(const v8h*)(Arow + k0g + kk + hh * 8);
      v8h ahi = *(const v8h*)(Arow + k0g + kk + 16 + hh * 8);
      v8h blo = *(const v8h*)(BrowBase + kk + hh * 8);
      v8h bhi = *(const v8h*)(BrowBase + kk + 16 + hh * 8);
      acc = wmma_f16(mk16(alo, ahi), mk16(blo, bhi), acc);
    }
    __syncthreads();
  }
  // epilogue: silu(acc/N) -> zbuf[b][s*20+c] (f16)
  const int o = nt * 16 + ln;
  if (o < C1){
#pragma unroll
    for (int r = 0; r < 8; ++r){
      const int m = mt * 16 + r + 8 * hh;
      const float zv = acc[r] * (1.f / (float)NN);
      zbuf[m * KTOT + s * C1 + o] = (_Float16)silu_f(zv);
    }
  }
}

// ---------------------------------------------------------------------------
// k2: hidden FilterMLP on v0 -> filt2^T [128][10240] f16.
// 16 group-samples per block.  GEMM1: 16x128 (K=96).  GEMM2: 16x2560 (K=128).
// ---------------------------------------------------------------------------
__global__ void __launch_bounds__(256)
k2_filt2(const float* __restrict__ v0,
         const _Float16* __restrict__ sw1T,
         const float* __restrict__ s_b1,
         const _Float16* __restrict__ sw2T,
         const float* __restrict__ s_b2,
         _Float16* __restrict__ fbufT){
  __shared__ _Float16 Ae[16][96];     // embed(v0), K-padded to 96
  __shared__ _Float16 H2[16][128];    // hidden activations (f16)
  const int s0  = blockIdx.x * 16;
  const int tid = threadIdx.x;

  for (int idx = tid; idx < 16 * 96; idx += 256){
    const int sl = idx / 96, k = idx % 96;
    float val = 0.f;
    if (k < 80){
      const int gd = k / 10, kk = k % 10;
      const float x = v0[(s0 + sl) * 8 + gd];
      const float f = (float)(1 << (kk % 5));
      val = (kk < 5) ? __sinf(x * f) : __cosf(x * f);
    }
    Ae[sl][k] = (_Float16)val;
  }
  __syncthreads();

  const int w = tid >> 5, lane = tid & 31;
  const int hh = lane >> 4, ln = lane & 15;

  // GEMM1: h2 = silu(v0e @ s_w1 + b1); wave w owns N-tile w
  {
    v8f acc = {};
    const _Float16* Brow = sw1T + (w * 16 + ln) * 96;
#pragma unroll
    for (int ks = 0; ks < 3; ++ks){
      const int kk = ks * 32;
      v8h alo = *(const v8h*)(&Ae[ln][kk + hh * 8]);
      v8h ahi = *(const v8h*)(&Ae[ln][kk + 16 + hh * 8]);
      v8h blo = *(const v8h*)(Brow + kk + hh * 8);
      v8h bhi = *(const v8h*)(Brow + kk + 16 + hh * 8);
      acc = wmma_f16(mk16(alo, ahi), mk16(blo, bhi), acc);
    }
    const int n = w * 16 + ln;
    const float bias = s_b1[n];
#pragma unroll
    for (int r = 0; r < 8; ++r) H2[r + 8 * hh][n] = (_Float16)silu_f(acc[r] + bias);
  }
  __syncthreads();

  // GEMM2: filt2 = h2 @ s_w2 + b2; wave w owns 20 N-tiles; A-frags hoisted
  v16h afrag[4];
#pragma unroll
  for (int ks = 0; ks < 4; ++ks){
    const int kk = ks * 32;
    v8h alo = *(const v8h*)(&H2[ln][kk + hh * 8]);
    v8h ahi = *(const v8h*)(&H2[ln][kk + 16 + hh * 8]);
    afrag[ks] = mk16(alo, ahi);
  }
  for (int t = w * 20; t < w * 20 + 20; ++t){
    const int jn = t * 16 + ln;                 // flattened (c*128 + w) column
    const _Float16* Brow = sw2T + jn * 128;
    v8f acc = {};
#pragma unroll
    for (int ks = 0; ks < 4; ++ks){
      const int kk = ks * 32;
      v8h blo = *(const v8h*)(Brow + kk + hh * 8);
      v8h bhi = *(const v8h*)(Brow + kk + 16 + hh * 8);
      acc = wmma_f16(afrag[ks], mk16(blo, bhi), acc);
    }
    const int c = jn >> 7, wc = jn & 127;
    const float bias = s_b2[jn];
#pragma unroll
    for (int r = 0; r < 8; ++r){
      const int srow = s0 + r + 8 * hh;
      fbufT[wc * KTOT + srow * C1 + c] = (_Float16)(acc[r] + bias);
    }
  }
}

// ---------------------------------------------------------------------------
// k3: z2[64][128] = zbuf[64,10240] @ filt2[10240,128] / S.
// 4 blocks x 8 waves = 32 tiles (4 M x 8 N). Both operands contiguous b128.
// ---------------------------------------------------------------------------
__global__ void __launch_bounds__(256)
k3_contract(const _Float16* __restrict__ zbuf,
            const _Float16* __restrict__ fbufT,
            float* __restrict__ z2){
  const int tid = threadIdx.x;
  const int w = tid >> 5, lane = tid & 31;
  const int hh = lane >> 4, ln = lane & 15;
  const int mt = w & 3, nt = blockIdx.x * 2 + (w >> 2);
  const _Float16* Arow = zbuf  + (mt * 16 + ln) * KTOT;
  const _Float16* Brow = fbufT + (nt * 16 + ln) * KTOT;
  v8f acc = {};
#pragma unroll 4
  for (int ks = 0; ks < KTOT / 32; ++ks){
    const int kk = ks * 32;
    if ((ks & 15) == 0){                         // stream-ahead prefetch hints
      __builtin_prefetch((const void*)(Arow + kk + 1024), 0, 0);
      __builtin_prefetch((const void*)(Brow + kk + 1024), 0, 0);
    }
    v8h alo = *(const v8h*)(Arow + kk + hh * 8);
    v8h ahi = *(const v8h*)(Arow + kk + 16 + hh * 8);
    v8h blo = *(const v8h*)(Brow + kk + hh * 8);
    v8h bhi = *(const v8h*)(Brow + kk + 16 + hh * 8);
    acc = wmma_f16(mk16(alo, ahi), mk16(blo, bhi), acc);
  }
#pragma unroll
  for (int r = 0; r < 8; ++r){
    const int m = mt * 16 + r + 8 * hh;
    const int n = nt * 16 + ln;
    z2[m * WIDTH + n] = acc[r] * (1.f / (float)SS);
  }
}

// ---------------------------------------------------------------------------
// k4: residual MLP tail + pool (fp32, one block, 32 KB LDS, in-place).
// ---------------------------------------------------------------------------
__global__ void __launch_bounds__(256)
k4_tail(const float* __restrict__ z2,
        const float* __restrict__ fc1_w, const float* __restrict__ fc1_b,
        const float* __restrict__ fc2_w, const float* __restrict__ fc2_b,
        const float* __restrict__ pool_w, const float* __restrict__ pool_b,
        float* __restrict__ out){
  __shared__ float Z[BB][WIDTH];
  const int tid = threadIdx.x;
  for (int idx = tid; idx < BB * WIDTH; idx += 256) Z[idx >> 7][idx & 127] = z2[idx];
  __syncthreads();
  float tmp[32];
  const float* Wt[2] = { fc1_w, fc2_w };
  const float* Bt[2] = { fc1_b, fc2_b };
  for (int layer = 0; layer < 2; ++layer){
    const float* Wl = Wt[layer]; const float* Bl = Bt[layer];
    int c = 0;
    for (int idx = tid; idx < BB * WIDTH; idx += 256, ++c){
      const int bb = idx >> 7, ww = idx & 127;
      float a = Bl[ww];
#pragma unroll 8
      for (int k = 0; k < WIDTH; ++k) a += Z[bb][k] * Wl[k * WIDTH + ww];
      tmp[c] = silu_f(a) + Z[bb][ww];
    }
    __syncthreads();
    c = 0;
    for (int idx = tid; idx < BB * WIDTH; idx += 256, ++c) Z[idx >> 7][idx & 127] = tmp[c];
    __syncthreads();
  }
  for (int idx = tid; idx < BB * 10; idx += 256){
    const int bb = idx / 10, o = idx % 10;
    float a = pool_b[o];
#pragma unroll 8
    for (int k = 0; k < WIDTH; ++k) a += Z[bb][k] * pool_w[k * 10 + o];
    out[idx] = a;
  }
}

// ---------------------------------------------------------------------------
extern "C" void kernel_launch(void* const* d_in, const int* in_sizes, int n_in,
                              void* d_out, int out_size, void* d_ws, size_t ws_size,
                              hipStream_t stream){
  (void)in_sizes; (void)n_in; (void)out_size; (void)ws_size;
  const float* data   = (const float*)d_in[0];
  const float* v0     = (const float*)d_in[1];
  const float* v_last = (const float*)d_in[2];
  const float* fl_w1  = (const float*)d_in[3];
  const float* fl_b1  = (const float*)d_in[4];
  const float* fl_w2  = (const float*)d_in[5];
  const float* fl_b2  = (const float*)d_in[6];
  const float* fl_w3  = (const float*)d_in[7];
  const float* fl_b3  = (const float*)d_in[8];
  const float* s_w1   = (const float*)d_in[9];
  const float* s_b1   = (const float*)d_in[10];
  const float* s_w2   = (const float*)d_in[11];
  const float* s_b2   = (const float*)d_in[12];
  const float* fc1_w  = (const float*)d_in[13];
  const float* fc1_b  = (const float*)d_in[14];
  const float* fc2_w  = (const float*)d_in[15];
  const float* fc2_b  = (const float*)d_in[16];
  const float* pool_w = (const float*)d_in[17];
  const float* pool_b = (const float*)d_in[18];

  char* ws = (char*)d_ws;
  _Float16* zdata = (_Float16*)(ws + OFF_ZDATA);
  _Float16* sw1T  = (_Float16*)(ws + OFF_SW1T);
  _Float16* sw2T  = (_Float16*)(ws + OFF_SW2T);
  _Float16* zbuf  = (_Float16*)(ws + OFF_ZBUF);
  _Float16* fbufT = (_Float16*)(ws + OFF_FBUFT);
  float*    z2    = (float*)(ws + OFF_Z2);
  float*    out   = (float*)d_out;

  k0_prep<<<(2560 * 128 + 255) / 256, 256, 0, stream>>>(data, s_w1, s_w2, zdata, sw1T, sw2T);
  k1_layer1<<<SS, 256, 0, stream>>>(zdata, v_last, fl_w1, fl_b1, fl_w2, fl_b2, fl_w3, fl_b3, zbuf);
  k2_filt2<<<SS / 16, 256, 0, stream>>>(v0, sw1T, s_b1, sw2T, s_b2, fbufT);
  k3_contract<<<4, 256, 0, stream>>>(zbuf, fbufT, z2);
  k4_tail<<<1, 256, 0, stream>>>(z2, fc1_w, fc1_b, fc2_w, fc2_b, pool_w, pool_b, out);
}
